// RNN_78391743087139
// MI455X (gfx1250) — matrix operand order repaired and checked by reading, not compiled, not run on hardware
//
#include <hip/hip_runtime.h>
#include <cstdint>

// ---------------------------------------------------------------------------
// Elman RNN (B=4096, T=512, I=1, H=32) on gfx1250.
//
// Per wave (16 batch rows, all T steps sequentially, h lives in VGPRs):
//   D[hid, batch](16x16 f32, 2 hid-tiles) =
//       W_hh(A frag, 16x32 f16) @ h(B frag, 32x16 f16)
//     + WMMA_F32_16X16X4_F32( [W_ih | bias] , [x_t | 1] )   // pre-activation
//   h_new = tanh(D);  y = h_new . W_out + b_out
//   D -> B fragment for next step: 8x cvt_pk_rtz + 8x shfl_xor(16) (no LDS).
// x load is software-pipelined one step ahead (loop is latency-bound).
// ---------------------------------------------------------------------------

typedef __attribute__((ext_vector_type(16))) _Float16 v16h;
typedef __attribute__((ext_vector_type(2)))  __fp16   h2;   // matches cvt_pkrtz return
typedef __attribute__((ext_vector_type(8)))  float    v8f;
typedef __attribute__((ext_vector_type(2)))  float    v2f;
typedef __attribute__((ext_vector_type(8)))  uint32_t v8u;

#define RNN_B 4096
#define RNN_T 512
#define RNN_H 32

#if __has_builtin(__builtin_amdgcn_wmma_f32_16x16x4_f32)
#define HAVE_WMMA4 1
#else
#define HAVE_WMMA4 0
#endif

__device__ __forceinline__ uint32_t pk16(float a, float b) {
  h2 p = __builtin_amdgcn_cvt_pkrtz(a, b);     // v_cvt_pk_rtz_f16_f32
  return __builtin_bit_cast(uint32_t, p);
}

__device__ __forceinline__ float fast_tanh(float v) {
#if __has_builtin(__builtin_amdgcn_tanhf)
  return __builtin_amdgcn_tanhf(v);            // v_tanh_f32 (TRANS, co-exec w/ WMMA)
#else
  return tanhf(v);
#endif
}

__global__ __launch_bounds__(32)
void rnn_elman_wmma(const float* __restrict__ x,      // [B,T,1]
                    const float* __restrict__ h0,     // [1,B,H]
                    const float* __restrict__ Wih,    // [H,1]
                    const float* __restrict__ Whh,    // [H,H]
                    const float* __restrict__ bih,    // [H]
                    const float* __restrict__ bhh,    // [H]
                    const float* __restrict__ Wout,   // [1,H]
                    const float* __restrict__ bout,   // [1]
                    float* __restrict__ outs,         // [B,T,1]
                    float* __restrict__ hlast)        // [1,B,H]
{
  const int lane = threadIdx.x;        // 0..31 (wave32)
  const int hi   = lane >> 4;          // half-wave: 0 = lanes 0-15, 1 = lanes 16-31
  const int lo16 = lane & 15;
  const int b    = blockIdx.x * 16 + lo16;   // batch row owned by this lane column
  const float* xrow = x + (size_t)b * RNN_T;

  // ---- W_hh as two f16 A-fragments (16x32, M = hid tile rows, K = hid in) ----
  // f16 A 16x32 k order used here (contiguous): lanes 0-15 hold M, V0..7 = K 0..15 (lo) / 16..31 (hi),
  // two K values packed per VGPR (matches the B fragment order).
  v16h aW[2];
  #pragma unroll
  for (int t = 0; t < 2; ++t) {
    const float* wrow = Whh + (size_t)(lo16 + 16 * t) * RNN_H;
    v8u u;
    #pragma unroll
    for (int v = 0; v < 8; ++v) {
      const int k0 = 2 * (v & 3) + 8 * (v >> 2) + 16 * hi;
      u[v] = pk16(wrow[k0], wrow[k0 + 1]);
    }
    aW[t] = __builtin_bit_cast(v16h, u);
  }

#if HAVE_WMMA4
  // ---- A fragment for the rank-2 pre-activation: c[m,n] = Wih[m]*x[n]+bias[m]
  // f32 A 16x4 layout: lanes 0-15 hold M, V0 = K0, V1 = K1; lanes 16-31 K2/K3 (=0).
  v2f aX[2];
  #pragma unroll
  for (int t = 0; t < 2; ++t) {
    const int m = lo16 + 16 * t;
    aX[t][0] = (hi == 0) ? Wih[m] : 0.0f;
    aX[t][1] = (hi == 0) ? (bih[m] + bhh[m]) : 0.0f;
  }
#else
  // VALU fallback: per-lane constants for c[r] = x*cw + cb
  float cwf[2][8], cbf[2][8];
  #pragma unroll
  for (int t = 0; t < 2; ++t)
    #pragma unroll
    for (int r = 0; r < 8; ++r) {
      const int m = 16 * t + r + 8 * hi;
      cwf[t][r] = Wih[m];
      cbf[t][r] = bih[m] + bhh[m];
    }
#endif

  // ---- output projection coefficients (per-lane scalars) ----
  float wo[2][8];
  #pragma unroll
  for (int t = 0; t < 2; ++t)
    #pragma unroll
    for (int r = 0; r < 8; ++r)
      wo[t][r] = Wout[16 * t + r + 8 * hi];
  const float bo = bout[0];

  // ---- initial hidden state as f16 B-fragment (32x16, K = hid, N = batch) ----
  // f16 B 32x16 k order used here (contiguous): lane = N column; V0..7 = K 0..15 (lo lanes) / 16..31 (hi).
  v16h hB;
  {
    const float* hrow = h0 + (size_t)b * RNN_H;
    v8u u;
    #pragma unroll
    for (int v = 0; v < 8; ++v) {
      const int k0 = 2 * (v & 3) + 8 * (v >> 2) + 16 * hi;
      u[v] = pk16(hrow[k0], hrow[k0 + 1]);
    }
    hB = __builtin_bit_cast(v16h, u);
  }

  float th[2][8];              // tanh'ed hidden values, kept for final-state write
  float xcur = xrow[0];        // software-pipelined x load (one step ahead)

  for (int t = 0; t < RNN_T; ++t) {
    // issue next step's x load now; its wait lands after a full step of work
    const int tn = (t + 1 < RNN_T) ? (t + 1) : (RNN_T - 1);   // clamped, no branch
    const float xnext = xrow[tn];
    __builtin_prefetch(&xrow[(t + 64 < RNN_T) ? (t + 64) : (RNN_T - 1)], 0, 0);

    // pre-activation tile c[m,n] = Wih[m]*x[n] + bias[m]
    v8f c0, c1;
#if HAVE_WMMA4
    v2f b4;
    b4[0] = (hi == 0) ? xcur : 0.0f;    // K0 row = x
    b4[1] = (hi == 0) ? 1.0f : 0.0f;    // K1 row = ones
    v8f zc = {};
    c0 = __builtin_amdgcn_wmma_f32_16x16x4_f32(false, aX[0], false, b4,
                                               (short)0, zc, false, false);
    c1 = __builtin_amdgcn_wmma_f32_16x16x4_f32(false, aX[1], false, b4,
                                               (short)0, zc, false, false);
#else
    #pragma unroll
    for (int r = 0; r < 8; ++r) {
      c0[r] = fmaf(xcur, cwf[0][r], cbf[0][r]);
      c1[r] = fmaf(xcur, cwf[1][r], cbf[1][r]);
    }
#endif

    // recurrent GEMM: D = W_hh @ h + c   (two 16-row hid tiles)
    v8f d0 = __builtin_amdgcn_wmma_f32_16x16x32_f16(false, aW[0], false, hB,
                                                    (short)0, c0, false, false);
    v8f d1 = __builtin_amdgcn_wmma_f32_16x16x32_f16(false, aW[1], false, hB,
                                                    (short)0, c1, false, false);

    // tanh + per-lane partial of the output projection
    float part = 0.0f;
    #pragma unroll
    for (int r = 0; r < 8; ++r) {
      th[0][r] = fast_tanh(d0[r]);
      th[1][r] = fast_tanh(d1[r]);
      part = fmaf(th[0][r], wo[0][r], part);
      part = fmaf(th[1][r], wo[1][r], part);
    }
    const float y = part + __shfl_xor(part, 16, 32) + bo;
    if (hi == 0) outs[(size_t)b * RNN_T + t] = y;

    // repack tanh(D) into the B fragment for the next step:
    //   V[v]   : lo = hid 2v,2v+1 (tile0)   | hi = hid 16+2v (tile1, from lo lanes)
    //   V[4+v] : lo = hid 8+2v (tile0, from hi lanes) | hi = hid 24+2v (tile1)
    v8u hu;
    #pragma unroll
    for (int v = 0; v < 4; ++v) {
      const uint32_t q0 = pk16(th[0][2 * v], th[0][2 * v + 1]);
      const uint32_t q1 = pk16(th[1][2 * v], th[1][2 * v + 1]);
      const uint32_t s0 = __shfl_xor(q0, 16, 32);
      const uint32_t s1 = __shfl_xor(q1, 16, 32);
      hu[v]     = (hi == 0) ? q0 : s1;
      hu[4 + v] = (hi == 0) ? s0 : q1;
    }
    hB = __builtin_bit_cast(v16h, hu);

    xcur = xnext;              // rotate the pipelined load
  }

  // ---- final hidden state [1,B,H] ----
  #pragma unroll
  for (int t = 0; t < 2; ++t)
    #pragma unroll
    for (int r = 0; r < 8; ++r)
      hlast[(size_t)b * RNN_H + 16 * t + 8 * hi + r] = th[t][r];
}

extern "C" void kernel_launch(void* const* d_in, const int* in_sizes, int n_in,
                              void* d_out, int out_size, void* d_ws, size_t ws_size,
                              hipStream_t stream) {
  (void)in_sizes; (void)n_in; (void)out_size; (void)d_ws; (void)ws_size;
  const float* x    = (const float*)d_in[0];
  const float* h0   = (const float*)d_in[1];
  const float* Wih  = (const float*)d_in[2];
  const float* Whh  = (const float*)d_in[3];
  const float* bih  = (const float*)d_in[4];
  const float* bhh  = (const float*)d_in[5];
  const float* Wout = (const float*)d_in[6];
  const float* bout = (const float*)d_in[7];

  float* outs  = (float*)d_out;                       // [B,T,1]
  float* hlast = outs + (size_t)RNN_B * RNN_T;        // [1,B,H]

  dim3 grid(RNN_B / 16);   // 256 single-wave workgroups -> spread across WGPs
  dim3 block(32);
  rnn_elman_wmma<<<grid, block, 0, stream>>>(x, h0, Wih, Whh, bih, bhh,
                                             Wout, bout, outs, hlast);
}
